// Attention_own_quan_52226802319927
// MI455X (gfx1250) — 8-bit matrix operand order repaired and checked by reading, compiled, not run on hardware
//
#include <hip/hip_runtime.h>
#include <stdint.h>


#define B_  4
#define S_  2048
#define D_  512
#define H_  8
#define DH_ 64
#define M_  (B_ * S_)

typedef __attribute__((ext_vector_type(8))) int v8i;

__device__ __forceinline__ v8i v8i_zero() {
  v8i z;
#pragma unroll
  for (int i = 0; i < 8; ++i) z[i] = 0;
  return z;
}

__device__ __forceinline__ v8i load_a_frag(const int8_t* rowbase, int half) {
  v8i f;
#pragma unroll
  for (int v = 0; v < 8; ++v) {
    const int koff = ((v >> 1) * 16) + half * 8 + ((v & 1) * 4);
    f[v] = *(const int*)(rowbase + koff);
  }
  return f;
}

__device__ __forceinline__ v8i load_b_frag(const int8_t* colbase, int half) {
  v8i f;
#pragma unroll
  for (int v = 0; v < 8; ++v) {
    const int koff = ((v >> 1) * 16) + half * 8 + ((v & 1) * 4);
    f[v] = *(const int*)(colbase + koff);
  }
  return f;
}

__device__ __forceinline__ float clampf(float x, float lo, float hi) {
  return fminf(fmaxf(x, lo), hi);
}
__device__ __forceinline__ int8_t q8(float x) {
  return (int8_t)(int)clampf(rintf(x), -128.f, 127.f);
}

__global__ void quant_act_kernel(const float* __restrict__ hs,
                                 const float* __restrict__ sc,
                                 int8_t* __restrict__ xq,
                                 int8_t* __restrict__ xk,
                                 int8_t* __restrict__ xv) {
  const int i  = blockIdx.x * blockDim.x + threadIdx.x;
  const float s0 = sc[0];
  const float hq = clampf(rintf(hs[i] / s0), -128.f, 127.f) * s0;
  xq[i] = q8(hq / sc[1]);
  xk[i] = q8(hq / sc[3]);
  xv[i] = q8(hq / sc[5]);
}

__global__ void quant_w_kernel(const float* __restrict__ W,
                               const float* __restrict__ sc, int iw,
                               int8_t* __restrict__ out) {
  const int i = blockIdx.x * blockDim.x + threadIdx.x;
  out[i] = q8(W[i] / sc[iw]);
}

__global__ __launch_bounds__(256) void gemm_qkv_kernel(
    const int8_t* __restrict__ X, const int8_t* __restrict__ W8,
    const float* __restrict__ sc, int ia, int iw, int io,
    int8_t* __restrict__ out, int vtrans) {
  const int wave = threadIdx.x >> 5;
  const int lane = threadIdx.x & 31;
  const int half = lane >> 4;
  const int l15  = lane & 15;

  const int tile = blockIdx.x * 8 + wave;
  const int mt = tile >> 5;
  const int nt = tile & 31;

  v8i acc = v8i_zero();
#pragma unroll
  for (int kk = 0; kk < 8; ++kk) {
    const int kb = kk * 64;
    const v8i a = load_a_frag(X  + (size_t)(mt * 16 + l15) * D_ + kb, half);
    const v8i b = load_b_frag(W8 + (size_t)(nt * 16 + l15) * D_ + kb, half);
    acc = __builtin_amdgcn_wmma_i32_16x16x64_iu8(true, a, true, b, acc,
                                                 false, false);
  }

  const float cscale = sc[ia] * sc[iw] / sc[io];
#pragma unroll
  for (int r = 0; r < 8; ++r) {
    const int m = mt * 16 + half * 8 + r;
    const int e = nt * 16 + l15;
    const int8_t v = q8((float)acc[r] * cscale);
    const int b_ = m >> 11;
    const int s_ = m & (S_ - 1);
    const int h_ = e >> 6;
    const int dh = e & (DH_ - 1);
    const int bh = b_ * H_ + h_;
    if (vtrans)
      out[((size_t)(bh * DH_ + dh)) * S_ + s_] = v;
    else
      out[((size_t)(bh * S_ + s_)) * DH_ + dh] = v;
  }
}

__global__ __launch_bounds__(128) void attn_kernel(
    const int8_t* __restrict__ q8p, const int8_t* __restrict__ k8p,
    const int8_t* __restrict__ vT8, const float* __restrict__ sc,
    int8_t* __restrict__ o8) {
  __shared__ float stile[4][16][16];
  __shared__ int   probs[4][16][16];
  __shared__ int   accbuf[4][4][16][16];
  __shared__ float mbuf[4][16], lbuf[4][16];
  __shared__ float Mrow[16], Linv[16];

  const int wave = threadIdx.x >> 5;
  const int lane = threadIdx.x & 31;
  const int half = lane >> 4;
  const int l15  = lane & 15;

  const int bh     = blockIdx.x >> 7;
  const int mblock = blockIdx.x & 127;
  const int b_ = bh >> 3, h_ = bh & 7;

  const float qkscale = sc[7] * sc[8] * 0.125f;
  const float s9 = sc[9], s10 = sc[10];

  const v8i aQ =
      load_a_frag(q8p + ((size_t)(bh * S_ + mblock * 16 + l15)) * DH_, half);

  float m_run = -1e30f, l_run = 0.f;
  for (int it = 0; it < 32; ++it) {
    const int jt = wave + it * 4;
    const v8i bK =
        load_b_frag(k8p + ((size_t)(bh * S_ + jt * 16 + l15)) * DH_, half);
    v8i sfrag = v8i_zero();
    sfrag = __builtin_amdgcn_wmma_i32_16x16x64_iu8(true, aQ, true, bK, sfrag,
                                                   false, false);
#pragma unroll
    for (int r = 0; r < 8; ++r)
      stile[wave][half * 8 + r][l15] = (float)sfrag[r] * qkscale;
    __syncthreads();
    if (lane < 16) {
      float tmax = stile[wave][lane][0];
#pragma unroll
      for (int c = 1; c < 16; ++c) tmax = fmaxf(tmax, stile[wave][lane][c]);
      const float mnew = fmaxf(m_run, tmax);
      float lsum = 0.f;
#pragma unroll
      for (int c = 0; c < 16; ++c) lsum += __expf(stile[wave][lane][c] - mnew);
      l_run = l_run * __expf(m_run - mnew) + lsum;
      m_run = mnew;
    }
    __syncthreads();
  }
  if (lane < 16) { mbuf[wave][lane] = m_run; lbuf[wave][lane] = l_run; }
  __syncthreads();
  if (threadIdx.x < 16) {
    const int r = threadIdx.x;
    float mf = mbuf[0][r];
#pragma unroll
    for (int w = 1; w < 4; ++w) mf = fmaxf(mf, mbuf[w][r]);
    float lf = 0.f;
#pragma unroll
    for (int w = 0; w < 4; ++w) lf += lbuf[w][r] * __expf(mbuf[w][r] - mf);
    Mrow[r] = mf;
    Linv[r] = 1.f / lf;
  }
  __syncthreads();

  v8i accN[4];
#pragma unroll
  for (int n = 0; n < 4; ++n) accN[n] = v8i_zero();

  for (int ci = 0; ci < 8; ++ci) {
    const int chunk = wave + ci * 4;
#pragma unroll
    for (int tt = 0; tt < 4; ++tt) {
      const int jt = chunk * 4 + tt;
      const v8i bK =
          load_b_frag(k8p + ((size_t)(bh * S_ + jt * 16 + l15)) * DH_, half);
      v8i sfrag = v8i_zero();
      sfrag = __builtin_amdgcn_wmma_i32_16x16x64_iu8(true, aQ, true, bK, sfrag,
                                                     false, false);
      unsigned char* pb = (unsigned char*)&probs[wave][0][0];
#pragma unroll
      for (int r = 0; r < 8; ++r) {
        const int row = half * 8 + r;
        const float p =
            __expf((float)sfrag[r] * qkscale - Mrow[row]) * Linv[row];
        const float u  = clampf(rintf(p / s9), 0.f, 255.f);
        const float u2 = clampf(rintf(u * s9 / s10), -128.f, 127.f);
        pb[row * 64 + tt * 16 + l15] = (unsigned char)(int)fmaxf(u2, 0.f);
      }
    }
    __syncthreads();
    const v8i aP = load_a_frag((const int8_t*)&probs[wave][l15][0], half);
#pragma unroll
    for (int ntile = 0; ntile < 4; ++ntile) {
      const v8i bV = load_b_frag(
          vT8 + ((size_t)(bh * DH_ + ntile * 16 + l15)) * S_ + chunk * 64,
          half);
      accN[ntile] = __builtin_amdgcn_wmma_i32_16x16x64_iu8(
          false  , aP, true  , bV, accN[ntile],
          false, false);
    }
    __syncthreads();
  }

#pragma unroll
  for (int n = 0; n < 4; ++n)
#pragma unroll
    for (int r = 0; r < 8; ++r)
      accbuf[wave][n][half * 8 + r][l15] = accN[n][r];
  __syncthreads();

  const float oscale = sc[10] * sc[11];
  const float s12inv = 1.f / sc[12];
#pragma unroll
  for (int r = 0; r < 8; ++r) {
    const int row = half * 8 + r;
    int sum = 0;
#pragma unroll
    for (int w2 = 0; w2 < 4; ++w2) sum += accbuf[w2][wave][row][l15];
    const int s_idx = mblock * 16 + row;
    const int d_idx = h_ * DH_ + wave * 16 + l15;
    o8[((size_t)(b_ * S_ + s_idx)) * D_ + d_idx] =
        q8((float)sum * oscale * s12inv);
  }
}

__global__ __launch_bounds__(256) void gemm_out_kernel(
    const int8_t* __restrict__ X, const int8_t* __restrict__ W8,
    const float* __restrict__ sc, const float* __restrict__ bias,
    float* __restrict__ out) {
  const int wave = threadIdx.x >> 5;
  const int lane = threadIdx.x & 31;
  const int half = lane >> 4;
  const int l15  = lane & 15;

  const int tile = blockIdx.x * 8 + wave;
  const int mt = tile >> 5;
  const int nt = tile & 31;

  v8i acc = v8i_zero();
#pragma unroll
  for (int kk = 0; kk < 8; ++kk) {
    const int kb = kk * 64;
    const v8i a = load_a_frag(X  + (size_t)(mt * 16 + l15) * D_ + kb, half);
    const v8i b = load_b_frag(W8 + (size_t)(nt * 16 + l15) * D_ + kb, half);
    acc = __builtin_amdgcn_wmma_i32_16x16x64_iu8(true, a, true, b, acc,
                                                 false, false);
  }

  const float cscale = sc[12] * sc[13];
#pragma unroll
  for (int r = 0; r < 8; ++r) {
    const int m = mt * 16 + half * 8 + r;
    const int e = nt * 16 + l15;
    out[(size_t)m * D_ + e] = (float)acc[r] * cscale + bias[e];
  }
}

extern "C" void kernel_launch(void* const* d_in, const int* in_sizes, int n_in,
                              void* d_out, int out_size, void* d_ws,
                              size_t ws_size, hipStream_t stream) {
  const float* hs = (const float*)d_in[0];
  const float* Wq = (const float*)d_in[1];
  const float* Wk = (const float*)d_in[2];
  const float* Wv = (const float*)d_in[3];
  const float* Wo = (const float*)d_in[4];
  const float* bo = (const float*)d_in[5];
  const float* sc = (const float*)d_in[6];
  float* out = (float*)d_out;

  char* ws = (char*)d_ws;
  const size_t ACT = (size_t)M_ * D_;
  const size_t WSZ = (size_t)D_ * D_;
  int8_t* xq  = (int8_t*)(ws + 0 * ACT);
  int8_t* xk  = (int8_t*)(ws + 1 * ACT);
  int8_t* xv  = (int8_t*)(ws + 2 * ACT);
  int8_t* wq8 = (int8_t*)(ws + 3 * ACT + 0 * WSZ);
  int8_t* wk8 = (int8_t*)(ws + 3 * ACT + 1 * WSZ);
  int8_t* wv8 = (int8_t*)(ws + 3 * ACT + 2 * WSZ);
  int8_t* wo8 = (int8_t*)(ws + 3 * ACT + 3 * WSZ);
  int8_t* q8b = (int8_t*)(ws + 3 * ACT + 4 * WSZ + 0 * ACT);
  int8_t* k8b = (int8_t*)(ws + 3 * ACT + 4 * WSZ + 1 * ACT);
  int8_t* vT8 = (int8_t*)(ws + 3 * ACT + 4 * WSZ + 2 * ACT);
  int8_t* o8b = (int8_t*)(ws + 3 * ACT + 4 * WSZ + 3 * ACT);

  quant_act_kernel<<<ACT / 256, 256, 0, stream>>>(hs, sc, xq, xk, xv);
  quant_w_kernel<<<WSZ / 256, 256, 0, stream>>>(Wq, sc, 2, wq8);
  quant_w_kernel<<<WSZ / 256, 256, 0, stream>>>(Wk, sc, 4, wk8);
  quant_w_kernel<<<WSZ / 256, 256, 0, stream>>>(Wv, sc, 6, wv8);
  quant_w_kernel<<<WSZ / 256, 256, 0, stream>>>(Wo, sc, 13, wo8);

  const int gemm_blocks = (M_ / 16) * (D_ / 16) / 8;
  gemm_qkv_kernel<<<gemm_blocks, 256, 0, stream>>>(xq, wq8, sc, 1, 2, 7,  q8b, 0);
  gemm_qkv_kernel<<<gemm_blocks, 256, 0, stream>>>(xk, wk8, sc, 3, 4, 8,  k8b, 0);
  gemm_qkv_kernel<<<gemm_blocks, 256, 0, stream>>>(xv, wv8, sc, 5, 6, 11, vT8, 1);

  attn_kernel<<<B_ * H_ * (S_ / 16), 128, 0, stream>>>(q8b, k8b, vT8, sc, o8b);

  gemm_out_kernel<<<gemm_blocks, 256, 0, stream>>>(o8b, wo8, sc, bo, out);
}
